// MultiWaveletProbAttention_26963804684339
// MI455X (gfx1250) — matrix operand order repaired and checked by reading, not compiled, not run on hardware
//
#include <hip/hip_runtime.h>
#include <stdint.h>

// Problem constants
#define B_   64
#define L_   256
#define H_   16
#define D_   64
#define C_   256
#define K_   8
#define NTOP 30
#define SAMPK 30
#define ICH  16384   // L*D
#define CK   2048    // C*K
#define M_DIM 1024   // B*H
#define SCALE_ 0.125f
#define NCHUNK (ICH / 32)

typedef __attribute__((ext_vector_type(16))) __bf16 v16bf;
typedef __attribute__((ext_vector_type(8)))  float  v8f;

// Exact packed conversion: one v_cvt_pk_bf16_f32 per dword, no half-shuffles.
static __device__ __forceinline__ uint32_t pack_bf2(float a, float b) {
    uint32_t r;
    asm("v_cvt_pk_bf16_f32 %0, %1, %2" : "=v"(r) : "v"(a), "v"(b));
    return r;
}
static __device__ __forceinline__ uint2 pack_bf4(float4 f) {
    uint2 r;
    r.x = pack_bf2(f.x, f.y);
    r.y = pack_bf2(f.z, f.w);
    return r;
}
static __device__ __forceinline__ v16bf load_frag(const uint16_t* lo, const uint16_t* hi) {
    union { v16bf v; uint4 u[2]; } r;
    r.u[0] = *reinterpret_cast<const uint4*>(lo);   // elements 0-7  -> VGPR0-3
    r.u[1] = *reinterpret_cast<const uint4*>(hi);   // elements 8-15 -> VGPR4-7
    return r.v;
}
static __device__ __forceinline__ v8f wmma_bf16(v16bf a, v16bf b, v8f c) {
    return __builtin_amdgcn_wmma_f32_16x16x32_bf16(false, a, false, b, (short)0, c, false, false);
}

// ---------------------------------------------------------------------------
// Kernel 1: wavelet GEMM  out[m,n] = sum_k X[m,k]*W[n,k] + bias[n]
//   m = b*H + h, k = l*D + d, X[m,k] = src[((b*L+l)*H+h)*D + d]
//   grid = (N/128, M/64, 2)   z: 0 -> (queries,Wq)->qw ; 1 -> (keys,Wk)->kw
//   Double-buffered LDS, software-pipelined float4 staging + cvt_pk_bf16.
// ---------------------------------------------------------------------------
#define XS 40   // padded LDS row stride (elements) for a 32-k chunk
__global__ __launch_bounds__(256)
void wavelet_gemm(const float* __restrict__ queries, const float* __restrict__ keys,
                  const float* __restrict__ Wq_w, const float* __restrict__ Wq_b,
                  const float* __restrict__ Wk_w, const float* __restrict__ Wk_b,
                  float* __restrict__ qw, float* __restrict__ kw)
{
    __shared__ __align__(16) uint16_t Xs[2][64 * XS];
    __shared__ __align__(16) uint16_t Ws[2][128 * XS];

    const int tid = threadIdx.x;
    const int sel = blockIdx.z;
    const float* Xg   = sel ? keys  : queries;
    const float* Wg   = sel ? Wk_w  : Wq_w;
    const float* bias = sel ? Wk_b  : Wq_b;
    float*       Og   = sel ? kw    : qw;

    const int blockN = blockIdx.x * 128;
    const int blockM = blockIdx.y * 64;

    const int lane  = tid & 31;
    const int wid   = tid >> 5;
    const int wm    = wid >> 2;       // 0..1
    const int wn    = wid & 3;        // 0..3
    const int hhalf = lane >> 4;      // 0..1
    const int rlan  = lane & 15;      // row / col within 16

    const int tx4 = tid & 7;          // k-quad index (k = tx4*4)
    const int ry  = tid >> 3;         // 0..31

    // Hoisted per-thread row base pointers (per-chunk offset is uniform/SALU).
    const float* xrow[2];
    #pragma unroll
    for (int p = 0; p < 2; ++p) {
        const int m  = blockM + p * 32 + ry;
        const int bb = m >> 4, hh = m & 15;
        xrow[p] = Xg + ((size_t)(bb * L_) * H_ + hh) * D_ + tx4 * 4;
    }
    const float* wrow[4];
    #pragma unroll
    for (int p = 0; p < 4; ++p)
        wrow[p] = Wg + (size_t)(blockN + p * 32 + ry) * ICH + tx4 * 4;

    v8f acc[2][2];
    #pragma unroll
    for (int i = 0; i < 2; ++i)
        #pragma unroll
        for (int j = 0; j < 2; ++j)
            acc[i][j] = v8f{0.f,0.f,0.f,0.f,0.f,0.f,0.f,0.f};

    float4 xf[2], wf[4];

    // prologue: chunk 0 -> buffer 0
    {
        #pragma unroll
        for (int p = 0; p < 2; ++p) xf[p] = *reinterpret_cast<const float4*>(xrow[p]);
        #pragma unroll
        for (int p = 0; p < 4; ++p) wf[p] = *reinterpret_cast<const float4*>(wrow[p]);
        #pragma unroll
        for (int p = 0; p < 2; ++p)
            *reinterpret_cast<uint2*>(&Xs[0][(p*32+ry)*XS + tx4*4]) = pack_bf4(xf[p]);
        #pragma unroll
        for (int p = 0; p < 4; ++p)
            *reinterpret_cast<uint2*>(&Ws[0][(p*32+ry)*XS + tx4*4]) = pack_bf4(wf[p]);
    }
    __syncthreads();

    for (int i = 0; i < NCHUNK; ++i) {
        const int s = i & 1;
        const bool more = (i + 1) < NCHUNK;

        // issue next chunk's global loads early (latency hidden by WMMA below)
        if (more) {
            const int kc = (i + 1) * 32;
            const size_t xoff = (size_t)(kc >> 6) * (H_ * D_) + (kc & 63);
            #pragma unroll
            for (int p = 0; p < 2; ++p) xf[p] = *reinterpret_cast<const float4*>(xrow[p] + xoff);
            #pragma unroll
            for (int p = 0; p < 4; ++p) {
                wf[p] = *reinterpret_cast<const float4*>(wrow[p] + kc);
                if (kc + 128 < ICH) __builtin_prefetch(wrow[p] + kc + 128, 0, 1); // global_prefetch
            }
        }

        // compute on buffer s (all fragment reads are contiguous ds_load_b128)
        {
            const uint16_t* xb = &Xs[s][(wm * 32 + rlan) * XS];
            v16bf a0 = load_frag(xb + hhalf * 8,            xb + 16 + hhalf * 8);
            v16bf a1 = load_frag(xb + 16 * XS + hhalf * 8,  xb + 16 * XS + 16 + hhalf * 8);
            const uint16_t* wb = &Ws[s][(wn * 32 + rlan) * XS];
            v16bf b0 = load_frag(wb + hhalf * 8,            wb + 16 + hhalf * 8);
            v16bf b1 = load_frag(wb + 16 * XS + hhalf * 8,  wb + 16 * XS + 16 + hhalf * 8);

            acc[0][0] = wmma_bf16(a0, b0, acc[0][0]);
            acc[0][1] = wmma_bf16(a0, b1, acc[0][1]);
            acc[1][0] = wmma_bf16(a1, b0, acc[1][0]);
            acc[1][1] = wmma_bf16(a1, b1, acc[1][1]);
        }

        // convert + store next chunk into the other buffer
        if (more) {
            const int d = s ^ 1;
            #pragma unroll
            for (int p = 0; p < 2; ++p)
                *reinterpret_cast<uint2*>(&Xs[d][(p*32+ry)*XS + tx4*4]) = pack_bf4(xf[p]);
            #pragma unroll
            for (int p = 0; p < 4; ++p)
                *reinterpret_cast<uint2*>(&Ws[d][(p*32+ry)*XS + tx4*4]) = pack_bf4(wf[p]);
        }
        __syncthreads();
    }

    // ---- store D: VGPR i -> row base + h*8 + i, col = base + (lane&15) ----
    #pragma unroll
    for (int tm = 0; tm < 2; ++tm) {
        #pragma unroll
        for (int tn = 0; tn < 2; ++tn) {
            const int row0 = blockM + wm * 32 + tm * 16 + hhalf * 8;
            const int col  = blockN + wn * 32 + tn * 16 + rlan;
            const float bv = bias[col];
            #pragma unroll
            for (int i = 0; i < 8; ++i)
                Og[(size_t)(row0 + i) * CK + col] = acc[tm][tn][i] + bv;
        }
    }
}

// ---------------------------------------------------------------------------
// Kernel 2: sampled QK measure M and top-30 selection. 1 block per (b,h).
// ---------------------------------------------------------------------------
__global__ __launch_bounds__(256)
void sample_topk(const float* __restrict__ qw, const float* __restrict__ kw,
                 const int* __restrict__ samp, int* __restrict__ idx_out)
{
    __shared__ float kws[CK];
    __shared__ float Ms[C_];
    const int bh  = blockIdx.x;
    const int tid = threadIdx.x;

    for (int i = tid; i < CK; i += 256) kws[i] = kw[(size_t)bh * CK + i];
    __syncthreads();

    {   // per-c measure
        const int c = tid;
        float q8[K_];
        #pragma unroll
        for (int j = 0; j < K_; ++j) q8[j] = qw[(size_t)bh * CK + c * K_ + j];
        float mx = -3.402823466e38f, sm = 0.f;
        for (int s = 0; s < SAMPK; ++s) {
            const int jc = samp[c * SAMPK + s];
            const float* kr = &kws[jc * K_];
            float d = 0.f;
            #pragma unroll
            for (int j = 0; j < K_; ++j) d += q8[j] * kr[j];
            mx = fmaxf(mx, d);
            sm += d;
        }
        Ms[c] = mx - sm * (1.0f / (float)C_);
    }
    __syncthreads();

    if (tid < 32) {   // wave 0: iterative argmax, values cached in registers
        const int base = tid * 8;
        float vals[8];
        #pragma unroll
        for (int t = 0; t < 8; ++t) vals[t] = Ms[base + t];
        for (int u = 0; u < NTOP; ++u) {
            float bv = -3.402823466e38f; int bi = -1;
            #pragma unroll
            for (int t = 0; t < 8; ++t)
                if (vals[t] > bv) { bv = vals[t]; bi = base + t; }
            #pragma unroll
            for (int off = 16; off >= 1; off >>= 1) {
                const float ov = __shfl_xor(bv, off, 32);
                const int   oi = __shfl_xor(bi, off, 32);
                if (ov > bv || (ov == bv && oi < bi)) { bv = ov; bi = oi; }
            }
            if (tid == 0) idx_out[bh * 32 + u] = bi;
            #pragma unroll
            for (int t = 0; t < 8; ++t)
                if (base + t == bi) vals[t] = -3.402823466e38f;
        }
    }
}

// ---------------------------------------------------------------------------
// Kernel 3: scores -> softmax -> upd ; cumsum(v) with scatter of upd rows.
// 1 block per (b,h).
// ---------------------------------------------------------------------------
__global__ __launch_bounds__(256)
void attn_out(const float* __restrict__ qw, const float* __restrict__ kw,
              const float* __restrict__ values, const int* __restrict__ idx,
              float* __restrict__ out)
{
    __shared__ float kws[CK];
    __shared__ float qreds[NTOP * K_];
    __shared__ int   idxs[NTOP];
    __shared__ int   lut[C_];
    __shared__ float sc[NTOP * C_];
    __shared__ float upds[NTOP * D_];
    __shared__ float qsums[4 * D_];

    const int bh  = blockIdx.x;
    const int b   = bh >> 4;
    const int h   = bh & 15;
    const int tid = threadIdx.x;

    for (int i = tid; i < CK; i += 256) kws[i] = kw[(size_t)bh * CK + i];
    if (tid < NTOP) idxs[tid] = idx[bh * 32 + tid];
    lut[tid] = -1;
    __syncthreads();
    if (tid < NTOP) lut[idxs[tid]] = tid;
    if (tid < NTOP * K_) {
        const int u = tid / K_, kkk = tid % K_;
        qreds[tid] = qw[(size_t)bh * CK + idxs[u] * K_ + kkk];
    }
    __syncthreads();

    // scores with causal mask (c > idx_u -> -inf)
    for (int e = tid; e < NTOP * C_; e += 256) {
        const int u = e >> 8, c = e & 255;
        const float* qr = &qreds[u * K_];
        const float* kr = &kws[c * K_];
        float s = 0.f;
        #pragma unroll
        for (int j = 0; j < K_; ++j) s += qr[j] * kr[j];
        s *= SCALE_;
        if (c > idxs[u]) s = -3.402823466e38f;
        sc[e] = s;
    }
    __syncthreads();

    // softmax: 8 lanes per row (groups aligned within a wave)
    {
        const int g = tid >> 3, j = tid & 7;
        if (g < NTOP) {
            float mx = -3.402823466e38f;
            for (int c = j; c < C_; c += 8) mx = fmaxf(mx, sc[g * C_ + c]);
            #pragma unroll
            for (int off = 4; off >= 1; off >>= 1)
                mx = fmaxf(mx, __shfl_xor(mx, off, 8));
            float sm = 0.f;
            for (int c = j; c < C_; c += 8) {
                const float e = __expf(sc[g * C_ + c] - mx);
                sc[g * C_ + c] = e;
                sm += e;
            }
            #pragma unroll
            for (int off = 4; off >= 1; off >>= 1)
                sm += __shfl_xor(sm, off, 8);
            const float inv = 1.0f / sm;
            for (int c = j; c < C_; c += 8) sc[g * C_ + c] *= inv;
        }
    }
    __syncthreads();

    // upd[u,d] = sum_c attn[u,c] * v[c,d]   (attn zero beyond idxs[u])
    for (int e = tid; e < NTOP * D_; e += 256) {
        const int u = e >> 6, d = e & 63;
        const int cmax = idxs[u];
        float a = 0.f;
        for (int c = 0; c <= cmax; ++c)
            a += sc[u * C_ + c] * values[(((size_t)(b * L_ + c) * H_ + h) * D_ + d)];
        upds[e] = a;
    }
    __syncthreads();

    // cumsum over l with scatter of upd rows
    {
        const int d = tid & 63, q = tid >> 6;
        float s = 0.f;
        for (int i = 0; i < 64; ++i)
            s += values[(((size_t)(b * L_ + (q * 64 + i)) * H_ + h) * D_ + d)];
        qsums[q * 64 + d] = s;
        __syncthreads();
        float acc = 0.f;
        for (int qq = 0; qq < q; ++qq) acc += qsums[qq * 64 + d];
        for (int i = 0; i < 64; ++i) {
            const int l = q * 64 + i;
            acc += values[(((size_t)(b * L_ + l) * H_ + h) * D_ + d)];
            const int u = lut[l];
            out[(((size_t)bh * L_ + l) * D_ + d)] = (u >= 0) ? upds[u * D_ + d] : acc;
        }
    }
}

// ---------------------------------------------------------------------------
extern "C" void kernel_launch(void* const* d_in, const int* in_sizes, int n_in,
                              void* d_out, int out_size, void* d_ws, size_t ws_size,
                              hipStream_t stream) {
    const float* queries = (const float*)d_in[0];
    const float* keys    = (const float*)d_in[1];
    const float* values  = (const float*)d_in[2];
    const float* Wq_w    = (const float*)d_in[3];
    const float* Wq_b    = (const float*)d_in[4];
    const float* Wk_w    = (const float*)d_in[5];
    const float* Wk_b    = (const float*)d_in[6];
    const int*   samp    = (const int*)d_in[9];
    float* out = (float*)d_out;

    float* qw  = (float*)d_ws;
    float* kw  = qw + (size_t)M_DIM * CK;
    int*   idx = (int*)(kw + (size_t)M_DIM * CK);

    dim3 g1(CK / 128, M_DIM / 64, 2);
    wavelet_gemm<<<g1, 256, 0, stream>>>(queries, keys, Wq_w, Wq_b, Wk_w, Wk_b, qw, kw);
    sample_topk<<<M_DIM, 256, 0, stream>>>(qw, kw, samp, idx);
    attn_out<<<M_DIM, 256, 0, stream>>>(qw, kw, values, idx, out);
}
